// DeformableTransformerEncoderLayer_876173328776
// MI455X (gfx1250) — compile-verified, matrix operand order checked by reading, not run on hardware
//
#include <hip/hip_runtime.h>

typedef __bf16 bf16;
typedef __attribute__((ext_vector_type(16))) __bf16 v16bf;
typedef __attribute__((ext_vector_type(8)))  __bf16 v8bf;
typedef __attribute__((ext_vector_type(8)))  float  v8f;

namespace {

constexpr int kS     = 13294;            // total spatial tokens per batch
constexpr int kBatch = 2;
constexpr int kM     = kBatch * kS;      // 26588 rows total
constexpr int kMPad  = 26624;            // round up to multiple of 64
constexpr int kC     = 256;              // d_model
constexpr int kFFN   = 1024;

// ---------------- workspace layout (bytes) ----------------
constexpr size_t SZ_QBF  = (size_t)kMPad * kC * 2;     // bf16 [Mpad,256]
constexpr size_t OFF_QBF = 0;
constexpr size_t OFF_SBF = OFF_QBF + SZ_QBF;
constexpr size_t SZ_OFFS = (size_t)kMPad * kC * 4;     // f32 [Mpad,256]
constexpr size_t OFF_OFFS = OFF_SBF + SZ_QBF;          // sampling offsets
constexpr size_t SZ_LOG  = (size_t)kMPad * 128 * 4;    // f32 [Mpad,128]
constexpr size_t OFF_LOG = OFF_OFFS + SZ_OFFS;         // attn logits -> softmax in place
constexpr size_t SZ_VAL  = (size_t)kMPad * kC * 4;
constexpr size_t OFF_VAL = OFF_LOG + SZ_LOG;           // value projection f32
constexpr size_t SZ_FF2  = (size_t)kMPad * kC * 4;
constexpr size_t OFF_FF2 = OFF_VAL + SZ_VAL;
constexpr size_t OFF_W   = OFF_FF2 + SZ_FF2;           // packed bf16 weights
constexpr size_t OFF_WOFF = OFF_W;                         // [256,256]
constexpr size_t OFF_WATT = OFF_WOFF + (size_t)256*256*2;  // [128,256]
constexpr size_t OFF_WVAL = OFF_WATT + (size_t)128*256*2;  // [256,256]
constexpr size_t OFF_WOUT = OFF_WVAL + (size_t)256*256*2;  // [256,256]
constexpr size_t OFF_W1   = OFF_WOUT + (size_t)256*256*2;  // [1024,256]
constexpr size_t OFF_W2   = OFF_W1   + (size_t)1024*256*2; // [256,1024]
// aliases (lifetimes do not overlap):
constexpr size_t OFF_OUTBF = OFF_QBF;   // bf16 [Mpad,256] attention output (q_bf dead)
constexpr size_t OFF_SRC2  = OFF_OFFS;  // f32  [Mpad,256] (offsets dead)
constexpr size_t OFF_XBF   = OFF_LOG;   // bf16 [Mpad,256] (aw dead)
constexpr size_t OFF_X     = OFF_VAL;   // f32  [Mpad,256] (value dead)
constexpr size_t OFF_FF1   = 0;         // bf16 [Mpad,1024] == exactly q+s+offs region

} // namespace

// ---------------- elementwise: q = bf16(src+pos), s = bf16(src) ----------------
__global__ void k_addcvt(const float* __restrict__ src, const float* __restrict__ pos,
                         bf16* __restrict__ q, bf16* __restrict__ s, int n) {
  int i = blockIdx.x * 256 + threadIdx.x;
  if (i < n) {
    float sv = src[i];
    q[i] = (bf16)(sv + pos[i]);
    s[i] = (bf16)sv;
  }
}

// ---------------- pack weight W[K,N] f32 (row major) -> Wt[N,K] bf16 ----------------
__global__ void k_wpack(const float* __restrict__ W, bf16* __restrict__ Wt, int K, int N) {
  int i = blockIdx.x * 256 + threadIdx.x;
  if (i < K * N) {
    int k = i / N, n = i - k * N;
    Wt[(size_t)n * K + k] = (bf16)W[i];
  }
}

// ---------------- bf16 WMMA GEMM:  C[M,N] = A[M,K] * Wt[N,K]^T + bias ----------------
// wave32: each wave computes a 32x32 output tile = 2x2 grid of 16x16 WMMA tiles
// (4 independent accumulator chains, A/B fragments each reused twice).
// Fragment layout (CDNA5 ISA, 16-bit A 16x32): lanes 0-15 hold K 0-7 & 16-23,
// lanes 16-31 hold K 8-15 & 24-31; B mirrors with N instead of M. Both A (row
// major [M,K]) and Wt (row major [N,K]) therefore load as two contiguous
// 16-byte chunks per lane.
template <bool RELU, bool OUTBF>
__global__ __launch_bounds__(256) void k_gemm(const bf16* __restrict__ A,
                                              const bf16* __restrict__ Bt,
                                              const float* __restrict__ bias,
                                              void* __restrict__ Cout,
                                              int N, int K) {
  const int lane = threadIdx.x & 31;
  const int wave = threadIdx.x >> 5;
  const int wm = wave & 1, wn = wave >> 1;          // 2x4 waves -> 64x128 block tile
  const int m0 = blockIdx.x * 64 + wm * 32;
  const int n0 = blockIdx.y * 128 + wn * 32;
  const int half = lane >> 4;                        // 0: K 0-7/16-23, 1: K 8-15/24-31
  const int l16 = lane & 15;
  const bf16* ap0 = A  + (size_t)(m0 + l16) * K + half * 8;
  const bf16* ap1 = ap0 + (size_t)16 * K;
  const bf16* bp0 = Bt + (size_t)(n0 + l16) * K + half * 8;
  const bf16* bp1 = bp0 + (size_t)16 * K;

  v8f acc00 = {0.f, 0.f, 0.f, 0.f, 0.f, 0.f, 0.f, 0.f};
  v8f acc01 = acc00, acc10 = acc00, acc11 = acc00;

  for (int k0 = 0; k0 < K; k0 += 32) {
    v8bf a0lo = *(const v8bf*)(ap0 + k0);
    v8bf a0hi = *(const v8bf*)(ap0 + k0 + 16);
    v8bf a1lo = *(const v8bf*)(ap1 + k0);
    v8bf a1hi = *(const v8bf*)(ap1 + k0 + 16);
    v8bf b0lo = *(const v8bf*)(bp0 + k0);
    v8bf b0hi = *(const v8bf*)(bp0 + k0 + 16);
    v8bf b1lo = *(const v8bf*)(bp1 + k0);
    v8bf b1hi = *(const v8bf*)(bp1 + k0 + 16);
    v16bf a0, a1, b0, b1;
#pragma unroll
    for (int e = 0; e < 8; ++e) {
      a0[e] = a0lo[e]; a0[e + 8] = a0hi[e];
      a1[e] = a1lo[e]; a1[e + 8] = a1hi[e];
      b0[e] = b0lo[e]; b0[e + 8] = b0hi[e];
      b1[e] = b1lo[e]; b1[e + 8] = b1hi[e];
    }
    acc00 = __builtin_amdgcn_wmma_f32_16x16x32_bf16(false, a0, false, b0,
                                                    (short)0, acc00, false, false);
    acc01 = __builtin_amdgcn_wmma_f32_16x16x32_bf16(false, a0, false, b1,
                                                    (short)0, acc01, false, false);
    acc10 = __builtin_amdgcn_wmma_f32_16x16x32_bf16(false, a1, false, b0,
                                                    (short)0, acc10, false, false);
    acc11 = __builtin_amdgcn_wmma_f32_16x16x32_bf16(false, a1, false, b1,
                                                    (short)0, acc11, false, false);
  }

  const int nA = n0 + l16;         // N column of tiles *0
  const int nB = n0 + 16 + l16;    // N column of tiles *1
  const float bnA = bias[nA];
  const float bnB = bias[nB];
#pragma unroll
  for (int v = 0; v < 8; ++v) {
    const int mA = m0 + v + half * 8;        // rows of tiles 0*
    const int mB = m0 + 16 + v + half * 8;   // rows of tiles 1*
    float v00 = acc00[v] + bnA;
    float v01 = acc01[v] + bnB;
    float v10 = acc10[v] + bnA;
    float v11 = acc11[v] + bnB;
    if (RELU) {
      v00 = fmaxf(v00, 0.f); v01 = fmaxf(v01, 0.f);
      v10 = fmaxf(v10, 0.f); v11 = fmaxf(v11, 0.f);
    }
    if (OUTBF) {
      bf16* C = (bf16*)Cout;
      C[(size_t)mA * N + nA] = (bf16)v00;
      C[(size_t)mA * N + nB] = (bf16)v01;
      C[(size_t)mB * N + nA] = (bf16)v10;
      C[(size_t)mB * N + nB] = (bf16)v11;
    } else {
      float* C = (float*)Cout;
      C[(size_t)mA * N + nA] = v00;
      C[(size_t)mA * N + nB] = v01;
      C[(size_t)mB * N + nA] = v10;
      C[(size_t)mB * N + nB] = v11;
    }
  }
}

// ---------------- softmax over 16 (levels*points) per (token, head), in place ----------------
__global__ void k_softmax(float* __restrict__ aw, int total) {
  int i = blockIdx.x * 256 + threadIdx.x;
  if (i >= total) return;
  float* p = aw + (size_t)i * 16;
  float mx = p[0];
#pragma unroll
  for (int j = 1; j < 16; ++j) mx = fmaxf(mx, p[j]);
  float e[16]; float s = 0.f;
#pragma unroll
  for (int j = 0; j < 16; ++j) { e[j] = __expf(p[j] - mx); s += e[j]; }
  float inv = 1.f / s;
#pragma unroll
  for (int j = 0; j < 16; ++j) p[j] = e[j] * inv;
}

// ---------------- deformable bilinear sampling + head-weighted sum ----------------
__device__ __forceinline__ float fetchv(const float* __restrict__ value, int vbase,
                                        int y, int x, int H, int W, int c) {
  if (y < 0 || y >= H || x < 0 || x >= W) return 0.f;
  return value[((size_t)(vbase + y * W + x)) * kC + c];
}

__global__ void k_sample(const float* __restrict__ off, const float* __restrict__ aw,
                         const float* __restrict__ value, const float* __restrict__ refp,
                         bf16* __restrict__ out) {
  const int m = blockIdx.x;            // token index [0, kM)
  const int c = threadIdx.x;           // channel [0, 256)
  const int n = m / kS;
  const int s = m - n * kS;
  const int head = c >> 5;

  const int Hs[4] = {100, 50, 25, 13};
  const int Ws[4] = {100, 50, 25, 13};
  const int St[4] = {0, 10000, 12500, 13125};

  const float* offp = off + (size_t)m * kC + head * 32;   // (lev,pt,2)
  const float* awp  = aw  + (size_t)m * 128 + head * 16;  // (lev,pt)
  const float* rp   = refp + (size_t)(n * kS + s) * 8;    // (lev,2)

  float acc = 0.f;
#pragma unroll
  for (int l = 0; l < 4; ++l) {
    const int Hl = Hs[l], Wl = Ws[l];
    const float rx = rp[l * 2 + 0], ry = rp[l * 2 + 1];
    const int vbase = n * kS + St[l];
#pragma unroll
    for (int p = 0; p < 4; ++p) {
      const float ox = offp[(l * 4 + p) * 2 + 0];
      const float oy = offp[(l * 4 + p) * 2 + 1];
      // loc*W - 0.5 == (ref + off/W)*W - 0.5
      const float x = rx * (float)Wl + ox - 0.5f;
      const float y = ry * (float)Hl + oy - 0.5f;
      const float x0f = floorf(x), y0f = floorf(y);
      const int x0 = (int)x0f, y0 = (int)y0f;
      const float wx1 = x - x0f, wx0 = 1.f - wx1;
      const float wy1 = y - y0f, wy0 = 1.f - wy1;
      const float a = awp[l * 4 + p];
      const float v00 = fetchv(value, vbase, y0,     x0,     Hl, Wl, c);
      const float v01 = fetchv(value, vbase, y0,     x0 + 1, Hl, Wl, c);
      const float v10 = fetchv(value, vbase, y0 + 1, x0,     Hl, Wl, c);
      const float v11 = fetchv(value, vbase, y0 + 1, x0 + 1, Hl, Wl, c);
      acc += a * (v00 * (wy0 * wx0) + v01 * (wy0 * wx1) +
                  v10 * (wy1 * wx0) + v11 * (wy1 * wx1));
    }
  }
  out[(size_t)m * kC + c] = (bf16)acc;
}

// ---------------- LayerNorm (one block of 256 threads per token) ----------------
__global__ void k_ln1(const float* __restrict__ src, const float* __restrict__ src2,
                      const float* __restrict__ g, const float* __restrict__ b,
                      float* __restrict__ x, bf16* __restrict__ xbf) {
  const int m = blockIdx.x, c = threadIdx.x;
  const size_t idx = (size_t)m * kC + c;
  float v = src[idx] + src2[idx];
  __shared__ float red[256];
  red[c] = v; __syncthreads();
  for (int o = 128; o > 0; o >>= 1) { if (c < o) red[c] += red[c + o]; __syncthreads(); }
  const float mean = red[0] * (1.f / kC);
  __syncthreads();
  const float d = v - mean;
  red[c] = d * d; __syncthreads();
  for (int o = 128; o > 0; o >>= 1) { if (c < o) red[c] += red[c + o]; __syncthreads(); }
  const float var = red[0] * (1.f / kC);
  const float r = rsqrtf(var + 1e-5f);
  const float o = d * r * g[c] + b[c];
  x[idx] = o;
  xbf[idx] = (bf16)o;
}

__global__ void k_ln2(const float* __restrict__ x, const float* __restrict__ ff,
                      const float* __restrict__ g, const float* __restrict__ b,
                      float* __restrict__ out) {
  const int m = blockIdx.x, c = threadIdx.x;
  const size_t idx = (size_t)m * kC + c;
  float v = x[idx] + ff[idx];
  __shared__ float red[256];
  red[c] = v; __syncthreads();
  for (int o = 128; o > 0; o >>= 1) { if (c < o) red[c] += red[c + o]; __syncthreads(); }
  const float mean = red[0] * (1.f / kC);
  __syncthreads();
  const float d = v - mean;
  red[c] = d * d; __syncthreads();
  for (int o = 128; o > 0; o >>= 1) { if (c < o) red[c] += red[c + o]; __syncthreads(); }
  const float var = red[0] * (1.f / kC);
  const float r = rsqrtf(var + 1e-5f);
  out[idx] = d * r * g[c] + b[c];
}

// ---------------- host-side launch sequence ----------------
extern "C" void kernel_launch(void* const* d_in, const int* in_sizes, int n_in,
                              void* d_out, int out_size, void* d_ws, size_t ws_size,
                              hipStream_t stream) {
  const float* src   = (const float*)d_in[0];
  const float* pos   = (const float*)d_in[1];
  const float* refp  = (const float*)d_in[2];
  // d_in[3] spatial_shapes, d_in[4] level_start_index: static metadata, unused
  const float* W_off = (const float*)d_in[5];
  const float* b_off = (const float*)d_in[6];
  const float* W_att = (const float*)d_in[7];
  const float* b_att = (const float*)d_in[8];
  const float* W_val = (const float*)d_in[9];
  const float* b_val = (const float*)d_in[10];
  const float* W_out = (const float*)d_in[11];
  const float* b_out = (const float*)d_in[12];
  const float* W1    = (const float*)d_in[13];
  const float* b1    = (const float*)d_in[14];
  const float* W2    = (const float*)d_in[15];
  const float* b2    = (const float*)d_in[16];
  const float* g1    = (const float*)d_in[17];
  const float* be1   = (const float*)d_in[18];
  const float* g2    = (const float*)d_in[19];
  const float* be2   = (const float*)d_in[20];

  char* ws = (char*)d_ws;
  bf16*  q_bf   = (bf16*)(ws + OFF_QBF);
  bf16*  s_bf   = (bf16*)(ws + OFF_SBF);
  float* offbuf = (float*)(ws + OFF_OFFS);
  float* awbuf  = (float*)(ws + OFF_LOG);
  float* valbuf = (float*)(ws + OFF_VAL);
  float* ff2buf = (float*)(ws + OFF_FF2);
  bf16*  WtOff  = (bf16*)(ws + OFF_WOFF);
  bf16*  WtAtt  = (bf16*)(ws + OFF_WATT);
  bf16*  WtVal  = (bf16*)(ws + OFF_WVAL);
  bf16*  WtOut  = (bf16*)(ws + OFF_WOUT);
  bf16*  Wt1    = (bf16*)(ws + OFF_W1);
  bf16*  Wt2    = (bf16*)(ws + OFF_W2);
  bf16*  out_bf = (bf16*)(ws + OFF_OUTBF);
  float* src2   = (float*)(ws + OFF_SRC2);
  bf16*  x_bf   = (bf16*)(ws + OFF_XBF);
  float* x_f    = (float*)(ws + OFF_X);
  bf16*  ff1_bf = (bf16*)(ws + OFF_FF1);

  const int nel = kM * kC;

  // 1) q = bf16(src+pos), s = bf16(src)
  k_addcvt<<<(nel + 255) / 256, 256, 0, stream>>>(src, pos, q_bf, s_bf, nel);

  // 2) pack weights (f32 [K,N] -> bf16 [N,K])
  k_wpack<<<(256 * 256 + 255) / 256, 256, 0, stream>>>(W_off, WtOff, 256, 256);
  k_wpack<<<(256 * 128 + 255) / 256, 256, 0, stream>>>(W_att, WtAtt, 256, 128);
  k_wpack<<<(256 * 256 + 255) / 256, 256, 0, stream>>>(W_val, WtVal, 256, 256);
  k_wpack<<<(256 * 256 + 255) / 256, 256, 0, stream>>>(W_out, WtOut, 256, 256);
  k_wpack<<<(256 * 1024 + 255) / 256, 256, 0, stream>>>(W1, Wt1, 256, 1024);
  k_wpack<<<(1024 * 256 + 255) / 256, 256, 0, stream>>>(W2, Wt2, 1024, 256);

  // 3-5) projection GEMMs (block tile 64x128)
  dim3 gOff(kMPad / 64, 256 / 128);
  k_gemm<false, false><<<gOff, 256, 0, stream>>>(q_bf, WtOff, b_off, offbuf, 256, 256);
  dim3 gAtt(kMPad / 64, 128 / 128);
  k_gemm<false, false><<<gAtt, 256, 0, stream>>>(q_bf, WtAtt, b_att, awbuf, 128, 256);
  k_gemm<false, false><<<gOff, 256, 0, stream>>>(s_bf, WtVal, b_val, valbuf, 256, 256);

  // 6) softmax over 16 per (token, head), in place
  k_softmax<<<(kM * 8 + 255) / 256, 256, 0, stream>>>(awbuf, kM * 8);

  // 7) deformable sampling -> bf16 attention output
  k_sample<<<kM, 256, 0, stream>>>(offbuf, awbuf, valbuf, refp, out_bf);

  // 8) output projection
  k_gemm<false, false><<<gOff, 256, 0, stream>>>(out_bf, WtOut, b_out, src2, 256, 256);

  // 9) residual + LayerNorm1
  k_ln1<<<kM, 256, 0, stream>>>(src, src2, g1, be1, x_f, x_bf);

  // 10) FFN up: relu(x @ W1 + b1) -> bf16
  dim3 gF1(kMPad / 64, 1024 / 128);
  k_gemm<true, true><<<gF1, 256, 0, stream>>>(x_bf, Wt1, b1, ff1_bf, 1024, 256);

  // 11) FFN down: ff1 @ W2 + b2
  k_gemm<false, false><<<gOff, 256, 0, stream>>>(ff1_bf, Wt2, b2, ff2buf, 256, 1024);

  // 12) residual + LayerNorm2 -> d_out
  k_ln2<<<kM, 256, 0, stream>>>(x_f, ff2buf, g2, be2, (float*)d_out);
}
